// Extractor_53017076302241
// MI455X (gfx1250) — matrix operand order repaired and checked by reading, not compiled, not run on hardware
//
#include <hip/hip_runtime.h>
#include <hip/hip_bf16.h>

// MI455X (gfx1250) per-pixel MLP: 6->32->128->128->32->16->1, SELU between.
// Compute-bound (~1800 FLOP/byte) => matmuls on v_wmma_f32_16x16x32_f16.
// Weights pre-swizzled into B-fragment lane layout in d_ws (contiguous 32B/lane),
// activations staged in per-wave LDS tiles (LDS is the lane transposer between
// C-fragment output layout and A-fragment input layout).
// Round-1 fixes: branchless SELU (no exec divergence), rolled nc-loops to keep
// VGPR count < 256 (kill s_set_vgpr_msb traffic, restore occupancy).

typedef __attribute__((ext_vector_type(16))) _Float16 v16h;
typedef __attribute__((ext_vector_type(8)))  float    v8f;

#define HW          65536      // 256*256
#define CH_STRIDE   65536
#define IMG_STRIDE  196608     // 3*65536

// d_ws fragment offsets in _Float16 elements (each chunk = 512 f16 = 1KB)
#define W1_OFF 0               // 32x128  -> 8 chunks
#define W2_OFF 4096            // 128x128 -> 32 chunks
#define W3_OFF 20480           // 128x32  -> 8 chunks
#define W4_OFF 24576           // 32x16   -> 1 chunk
#define WS_F16_TOTAL 25088     // 50,176 bytes

#define WAVES 4
#define TILES_PER_WAVE 4

// ---------------- weight pre-swizzle (once per launch, tiny) ----------------
// B-operand layout per CDNA5 ISA (16-bit B, 32x16 per chunk):
//   lane l<16 : N=l,      VGPR0-3 K=0..7,  VGPR4-7 K=16..23
//   lane l>=16: N=l-16,   VGPR0-3 K=8..15, VGPR4-7 K=24..31
// Stored so each lane's 8 VGPRs are 32 contiguous bytes at chunkBase + l*32.
__device__ __forceinline__ void swizzle_store(const float* W, _Float16* dst,
                                              int K, int N, int tid, int nth) {
  int total = K * N;
  int ntiles = N >> 4;
  for (int e = tid; e < total; e += nth) {
    int k = e / N, n = e % N;
    int kc = k >> 5, nc = n >> 4;
    int kk = k & 31, nn = n & 15;
    int lane = nn + ((kk & 8) ? 16 : 0);
    int v = ((kk & 7) >> 1) + ((kk & 16) ? 4 : 0);
    int h = kk & 1;
    int chunk = kc * ntiles + nc;
    dst[chunk * 512 + lane * 16 + v * 2 + h] = (_Float16)W[e];
  }
}

__global__ void prep_weights_kernel(const float* __restrict__ W1,
                                    const float* __restrict__ W2,
                                    const float* __restrict__ W3,
                                    const float* __restrict__ W4,
                                    _Float16* __restrict__ wf) {
  int tid = blockIdx.x * blockDim.x + threadIdx.x;
  int nth = gridDim.x * blockDim.x;
  swizzle_store(W1, wf + W1_OFF, 32, 128, tid, nth);
  swizzle_store(W2, wf + W2_OFF, 128, 128, tid, nth);
  swizzle_store(W3, wf + W3_OFF, 128, 32, tid, nth);
  swizzle_store(W4, wf + W4_OFF, 32, 16, tid, nth);
}

// ---------------- main kernel helpers ----------------
union AFrag { unsigned u[8]; v16h h; };
union BFrag { uint4 q[2]; v16h h; };

// Branchless SELU: lam*max(x,0) + lam*alpha*(exp(min(x,0)) - 1).
// For x>0 the exp term is exp(0)-1 == 0; for x<=0 the linear term is 0.
// No compare/select, no exec-mask divergence: max, min, exp, fma chain only.
__device__ __forceinline__ float selu_f(float x) {
  const float lam       = 1.0507009873554805f;
  const float lam_alpha = 1.7580993408473766f;  // lam * 1.6732632423543772
  float p = fmaxf(x, 0.0f);
  float n = fminf(x, 0.0f);
  return __builtin_fmaf(lam, p, lam_alpha * (__expf(n) - 1.0f));
}

// A-operand per CDNA5 ISA (16-bit A, 16x32): lane l, M=l%16;
//   lanes<16 : VGPR0-3 K=0..7,  VGPR4-7 K=16..23
//   lanes>=16: VGPR0-3 K=8..15, VGPR4-7 K=24..31
__device__ __forceinline__ AFrag load_a(const _Float16* act, int stride, int k0, int lane) {
  AFrag a;
  int m  = lane & 15;
  int hs = (lane >> 4) & 1;
  const _Float16* row = act + m * stride + k0 + hs * 8;
#pragma unroll
  for (int v = 0; v < 4; ++v) __builtin_memcpy(&a.u[v],     row + 2 * v,      4);
#pragma unroll
  for (int v = 0; v < 4; ++v) __builtin_memcpy(&a.u[4 + v], row + 16 + 2 * v, 4);
  return a;
}

__device__ __forceinline__ BFrag load_b(const _Float16* wfrag, int chunk, int lane) {
  BFrag b;
  __builtin_memcpy(&b, wfrag + chunk * 512 + lane * 16, 32);  // 2x global_load_b128
  return b;
}

// C/D fragment: lanes<16: N=lane, M=r; lanes>=16: N=lane-16, M=8+r.
// Fused bias + SELU, store f16 row-major to next activation buffer.
__device__ __forceinline__ void store_tile(v8f acc, const float* bias, int n0,
                                           _Float16* out, int stride, int lane) {
  int col = n0 * 16 + (lane & 15);
  int hs  = (lane >> 4) & 1;
  float bv = bias[col];
  _Float16* o = out + (hs * 8) * stride + col;
#pragma unroll
  for (int r = 0; r < 8; ++r)
    o[r * stride] = (_Float16)selu_f(acc[r] + bv);
}

#define WMMA_F16(A, B, C) \
  __builtin_amdgcn_wmma_f32_16x16x32_f16(false, (A), false, (B), (short)0, (C), false, false)

// ---------------- main MLP kernel ----------------
__global__ __launch_bounds__(128) void mlp_kernel(
    const float* __restrict__ x, const float* __restrict__ y,
    const float* __restrict__ W0,
    const float* __restrict__ b0, const float* __restrict__ b1,
    const float* __restrict__ b2, const float* __restrict__ b3,
    const float* __restrict__ b4,
    const float* __restrict__ W5, const float* __restrict__ b5,
    const _Float16* __restrict__ wf, float* __restrict__ out) {
  __shared__ __align__(16) _Float16 acts[WAVES][2][2048];  // 2x 16x128 f16 per wave
  __shared__ float sW0[192];    // 6x32
  __shared__ float sBias[337];  // b0(32) b1(128) b2(128) b3(32) b4(16) b5(1)
  __shared__ float sW5[16];

  int tid = threadIdx.x;
  for (int i = tid; i < 192; i += 128) sW0[i] = W0[i];
  for (int i = tid; i < 337; i += 128) {
    float v;
    if      (i < 32)  v = b0[i];
    else if (i < 160) v = b1[i - 32];
    else if (i < 288) v = b2[i - 160];
    else if (i < 320) v = b3[i - 288];
    else if (i < 336) v = b4[i - 320];
    else              v = b5[0];
    sBias[i] = v;
  }
  if (tid < 16) sW5[tid] = W5[tid];
  __syncthreads();

  int wave = tid >> 5;
  int lane = tid & 31;
  _Float16* buf0 = acts[wave][0];
  _Float16* buf1 = acts[wave][1];
  const float* sb1 = sBias + 32;
  const float* sb2 = sBias + 160;
  const float* sb3 = sBias + 288;
  const float* sb4 = sBias + 320;

  int m    = lane & 15;
  int half = (lane >> 4) & 1;

  for (int it = 0; it < TILES_PER_WAVE; ++it) {
    int tile      = (blockIdx.x * WAVES + wave) * TILES_PER_WAVE + it;
    int tokenBase = tile << 4;                 // 16 tokens, same batch image
    int bb        = tokenBase >> 16;
    int p         = tokenBase & (HW - 1);

    const float* xr = x + (size_t)bb * IMG_STRIDE + p + m;
    const float* yr = y + (size_t)bb * IMG_STRIDE + p + m;

    if (it + 1 < TILES_PER_WAVE) {             // prefetch next tile's inputs
      __builtin_prefetch(xr + 16, 0, 0);
      __builtin_prefetch(yr + 16, 0, 0);
    }

    // ---- layer 0: 6 -> 32, VALU (K=6 too small for WMMA) ----
    float in6[6];
    in6[0] = xr[0]; in6[1] = xr[CH_STRIDE]; in6[2] = xr[2 * CH_STRIDE];
    in6[3] = yr[0]; in6[4] = yr[CH_STRIDE]; in6[5] = yr[2 * CH_STRIDE];
    {
      _Float16* row = buf0 + m * 32 + half * 16;
#pragma unroll
      for (int j = 0; j < 16; ++j) {
        int col = half * 16 + j;
        float acc = sBias[col];
#pragma unroll
        for (int c = 0; c < 6; ++c) acc += in6[c] * sW0[c * 32 + col];
        row[j] = (_Float16)selu_f(acc);
      }
    }

    // ---- layer 1: 32 -> 128 (8 WMMA) ----
    {
      AFrag a = load_a(buf0, 32, 0, lane);
      for (int nc = 0; nc < 8; ++nc) {         // rolled: keep VGPR pressure low
        BFrag b = load_b(wf + W1_OFF, nc, lane);
        v8f acc = {};
        acc = WMMA_F16(a.h, b.h, acc);
        store_tile(acc, sb1, nc, buf1, 128, lane);
      }
    }

    // ---- layer 2: 128 -> 128 (32 WMMA) ----
    {
      AFrag a[4];
#pragma unroll
      for (int kc = 0; kc < 4; ++kc) a[kc] = load_a(buf1, 128, kc * 32, lane);
      for (int nc = 0; nc < 8; ++nc) {         // rolled
        v8f acc = {};
#pragma unroll
        for (int kc = 0; kc < 4; ++kc) {
          BFrag b = load_b(wf + W2_OFF, kc * 8 + nc, lane);
          acc = WMMA_F16(a[kc].h, b.h, acc);
        }
        store_tile(acc, sb2, nc, buf0, 128, lane);
      }
    }

    // ---- layer 3: 128 -> 32 (8 WMMA) ----
    {
      AFrag a[4];
#pragma unroll
      for (int kc = 0; kc < 4; ++kc) a[kc] = load_a(buf0, 128, kc * 32, lane);
      for (int nc = 0; nc < 2; ++nc) {         // rolled
        v8f acc = {};
#pragma unroll
        for (int kc = 0; kc < 4; ++kc) {
          BFrag b = load_b(wf + W3_OFF, kc * 2 + nc, lane);
          acc = WMMA_F16(a[kc].h, b.h, acc);
        }
        store_tile(acc, sb3, nc, buf1, 32, lane);
      }
    }

    // ---- layer 4: 32 -> 16 (1 WMMA) ----
    {
      AFrag a = load_a(buf1, 32, 0, lane);
      BFrag b = load_b(wf + W4_OFF, 0, lane);
      v8f acc = {};
      acc = WMMA_F16(a.h, b.h, acc);
      store_tile(acc, sb4, 0, buf0, 16, lane);
    }

    // ---- layer 5: 16 -> 1, VALU dot, no activation ----
    {
      const _Float16* r5 = buf0 + m * 16;
      float s = sBias[336];
#pragma unroll
      for (int k = 0; k < 16; ++k) s += (float)r5[k] * sW5[k];
      if (lane < 16) out[tokenBase + m] = s;
    }
  }
}

// ---------------- launch ----------------
extern "C" void kernel_launch(void* const* d_in, const int* in_sizes, int n_in,
                              void* d_out, int out_size, void* d_ws, size_t ws_size,
                              hipStream_t stream) {
  (void)in_sizes; (void)n_in; (void)out_size; (void)ws_size;
  const float* x  = (const float*)d_in[0];
  const float* y  = (const float*)d_in[1];
  const float* W0 = (const float*)d_in[2];
  const float* b0 = (const float*)d_in[3];
  const float* W1 = (const float*)d_in[4];
  const float* b1 = (const float*)d_in[5];
  const float* W2 = (const float*)d_in[6];
  const float* b2 = (const float*)d_in[7];
  const float* W3 = (const float*)d_in[8];
  const float* b3 = (const float*)d_in[9];
  const float* W4 = (const float*)d_in[10];
  const float* b4 = (const float*)d_in[11];
  const float* W5 = (const float*)d_in[12];
  const float* b5 = (const float*)d_in[13];
  _Float16* wf = (_Float16*)d_ws;

  prep_weights_kernel<<<64, 256, 0, stream>>>(W1, W2, W3, W4, wf);

  // 1,048,576 tokens / 16 per tile = 65,536 tiles = 4096 blocks * 4 waves * 4 tiles
  mlp_kernel<<<4096, 128, 0, stream>>>(x, y, W0, b0, b1, b2, b3, b4, W5, b5,
                                       wf, (float*)d_out);
}
